// LowRankGaussian_85684597555236
// MI455X (gfx1250) — matrix operand order repaired and checked by reading, not compiled, not run on hardware
//
#include <hip/hip_runtime.h>

#define N_HIDDEN 512
#define N_LAT    128
#define N_COLS   768   // 128 (mu) + 512 (u) + 128 (d)
#define ROWS_PB  32    // 2 row tiles of 16 -> register-level B reuse
#define SX_STRIDE 520  // 8-element pad -> bank spread for ds_read_b128
#define SC_STRIDE 772  // 768 + 4 pad

typedef __bf16 v16bf __attribute__((ext_vector_type(16)));
typedef __bf16 v8bf  __attribute__((ext_vector_type(8)));
typedef float  v8f   __attribute__((ext_vector_type(8)));

union BF16x16 { v16bf v; v8bf h[2]; };

// ---------------------------------------------------------------------------
// Kernel 1: pack W_mu | W_u | W_d (fp32, [512 x C] row-major) into a single
// transposed bf16 matrix Wt[768][512] so B-fragment loads are contiguous.
// ---------------------------------------------------------------------------
__global__ void lrg_build_wt(const float* __restrict__ Wmu,
                             const float* __restrict__ Wu,
                             const float* __restrict__ Wd,
                             __bf16* __restrict__ Wt) {
    int idx = blockIdx.x * 256 + threadIdx.x;      // over 768*512
    if (idx >= N_COLS * N_HIDDEN) return;
    int c = idx >> 9;       // output column 0..767 (row of Wt)
    int h = idx & 511;      // hidden index 0..511
    float v;
    if (c < 128)      v = Wmu[h * 128 + c];
    else if (c < 640) v = Wu [h * 512 + (c - 128)];
    else              v = Wd [h * 128 + (c - 640)];
    Wt[idx] = (__bf16)v;
}

__device__ __forceinline__ float lrg_red8(float v) {
    #pragma unroll
    for (int off = 4; off; off >>= 1) v += __shfl_xor(v, off, 8);
    return v;
}

// ---------------------------------------------------------------------------
// Kernel 2: fused  [32 x 512] @ [512 x 768]  +  per-row KL epilogue.
// Each of 8 waves owns 6 column tiles; each B fragment feeds 2 row tiles.
// ---------------------------------------------------------------------------
__global__ void __launch_bounds__(256, 2)
lrg_fused_kl(const float* __restrict__ x,
             const float* __restrict__ bmu,
             const float* __restrict__ bu,
             const float* __restrict__ bd,
             const __bf16* __restrict__ Wt,
             float* __restrict__ partials) {
    __shared__ __bf16 sx[ROWS_PB * SX_STRIDE];
    __shared__ float  sc[ROWS_PB * SC_STRIDE];
    __shared__ float  skl[ROWS_PB];

    const int tid  = threadIdx.x;
    const int wave = tid >> 5;
    const int lane = tid & 31;
    const int m    = lane & 15;
    const int half = lane >> 4;
    const int row0 = blockIdx.x * ROWS_PB;

    // ---- Phase 1: x tile (32x512 f32) -> bf16 LDS ----
    #pragma unroll
    for (int i = 0; i < 16; ++i) {
        int f    = tid + i * 256;          // float4 index over 32*128
        int row  = f >> 7;
        int col4 = f & 127;
        float4 val = ((const float4*)(x + (size_t)(row0 + row) * N_HIDDEN))[col4];
        __bf16* dst = &sx[row * SX_STRIDE + col4 * 4];
        dst[0] = (__bf16)val.x; dst[1] = (__bf16)val.y;
        dst[2] = (__bf16)val.z; dst[3] = (__bf16)val.w;
    }
    __syncthreads();

    // ---- Phase 2: WMMA GEMM ----
    v8f acc0[6] = {};   // row tile 0 (rows 0..15)
    v8f acc1[6] = {};   // row tile 1 (rows 16..31)
    // B fragment: lane holds column n=m, chunks K = half*8+0..7 and 16+half*8+0..7.
    const __bf16* wbase = Wt + ((size_t)(wave * 6 * 16 + m) * N_HIDDEN) + half * 8;

    for (int kt = 0; kt < 16; ++kt) {
        // A fragments: row (rt*16+m), chunks K = half*8+0..7 and 16+half*8+0..7
        BF16x16 a0, a1;
        const __bf16* ap0 = &sx[m * SX_STRIDE + kt * 32 + half * 8];
        const __bf16* ap1 = ap0 + 16 * SX_STRIDE;
        a0.h[0] = *(const v8bf*)ap0;
        a0.h[1] = *(const v8bf*)(ap0 + 16);
        a1.h[0] = *(const v8bf*)ap1;
        a1.h[1] = *(const v8bf*)(ap1 + 16);
        #pragma unroll
        for (int j = 0; j < 6; ++j) {
            const __bf16* bp = wbase + (size_t)j * 16 * N_HIDDEN + kt * 32;
            __builtin_prefetch(bp + 32, 0, 3);   // next K-step of this B stream
            BF16x16 b;
            b.h[0] = *(const v8bf*)bp;
            b.h[1] = *(const v8bf*)(bp + 16);
            acc0[j] = __builtin_amdgcn_wmma_f32_16x16x32_bf16(
                false, a0.v, false, b.v, (short)0, acc0[j], false, false);
            acc1[j] = __builtin_amdgcn_wmma_f32_16x16x32_bf16(
                false, a1.v, false, b.v, (short)0, acc1[j], false, false);
        }
    }

    // ---- Phase 3: accumulators (+bias) -> LDS C tile ----
    #pragma unroll
    for (int j = 0; j < 6; ++j) {
        int c = (wave * 6 + j) * 16 + m;   // output column, lane-constant over v
        float bias = (c < 128) ? bmu[c] : ((c < 640) ? bu[c - 128] : bd[c - 640]);
        #pragma unroll
        for (int v = 0; v < 8; ++v) {
            sc[(half * 8 + v) * SC_STRIDE + c]        = acc0[j][v] + bias;
            sc[(16 + half * 8 + v) * SC_STRIDE + c]   = acc1[j][v] + bias;
        }
    }
    __syncthreads();

    // ---- Phase 4: per-row KL. 8 threads per row, 16 latents each. ----
    const int row = tid >> 3;
    const int t   = tid & 7;
    const float* crow = &sc[row * SC_STRIDE];

    float musq = 0.f, sd = 0.f, st = 0.f, su2 = 0.f;
    float m00=0,m01=0,m02=0,m03=0,m11=0,m12=0,m13=0,m22=0,m23=0,m33=0;
    #pragma unroll
    for (int i = 0; i < 16; ++i) {
        int l = t + i * 8;
        float mu = crow[l];                 musq += mu * mu;
        float tl = crow[640 + l];           // pre-activation: log d
        float dl  = __expf(tl);
        float inv = __expf(-tl);            // 1/d
        sd += dl; st += tl;
        float u0 = crow[128 + l * 4 + 0];
        float u1 = crow[128 + l * 4 + 1];
        float u2 = crow[128 + l * 4 + 2];
        float u3 = crow[128 + l * 4 + 3];
        su2 += u0*u0 + u1*u1 + u2*u2 + u3*u3;
        m00 += u0*u0*inv; m01 += u0*u1*inv; m02 += u0*u2*inv; m03 += u0*u3*inv;
        m11 += u1*u1*inv; m12 += u1*u2*inv; m13 += u1*u3*inv;
        m22 += u2*u2*inv; m23 += u2*u3*inv; m33 += u3*u3*inv;
    }
    musq = lrg_red8(musq); sd = lrg_red8(sd); st = lrg_red8(st); su2 = lrg_red8(su2);
    m00 = lrg_red8(m00); m01 = lrg_red8(m01); m02 = lrg_red8(m02); m03 = lrg_red8(m03);
    m11 = lrg_red8(m11); m12 = lrg_red8(m12); m13 = lrg_red8(m13);
    m22 = lrg_red8(m22); m23 = lrg_red8(m23); m33 = lrg_red8(m33);

    if (t == 0) {
        // det(I4 + M^T M), symmetric SPD -> sum(log1p(eigvals)) == log(det)
        float a00 = 1.f + m00, a11 = 1.f + m11, a22 = 1.f + m22, a33 = 1.f + m33;
        float M0 = a11*(a22*a33 - m23*m23) - m12*(m12*a33 - m23*m13) + m13*(m12*m23 - a22*m13);
        float M1 = m01*(a22*a33 - m23*m23) - m12*(m02*a33 - m23*m03) + m13*(m02*m23 - a22*m03);
        float M2 = m01*(m12*a33 - m23*m13) - a11*(m02*a33 - m23*m03) + m13*(m02*m13 - m12*m03);
        float M3 = m01*(m12*m23 - a22*m13) - a11*(m02*m23 - a22*m03) + m12*(m02*m13 - m12*m03);
        float det = a00*M0 - m01*M1 + m02*M2 - m03*M3;
        skl[row] = 0.5f * (sd + su2 + musq - (float)N_LAT - st - __logf(det));
    }
    __syncthreads();

    if (tid == 0) {
        float s = 0.f;
        #pragma unroll
        for (int r = 0; r < ROWS_PB; ++r) s += skl[r];   // fixed order: deterministic
        partials[blockIdx.x] = s;
    }
}

// ---------------------------------------------------------------------------
// Kernel 3: deterministic final reduction of per-block partials.
// ---------------------------------------------------------------------------
__global__ void lrg_reduce(const float* __restrict__ partials, int nblocks,
                           float* __restrict__ out, float invN) {
    __shared__ float s[256];
    float acc = 0.f;
    for (int i = threadIdx.x; i < nblocks; i += 256) acc += partials[i];
    s[threadIdx.x] = acc;
    __syncthreads();
    for (int w = 128; w; w >>= 1) {
        if (threadIdx.x < w) s[threadIdx.x] += s[threadIdx.x + w];
        __syncthreads();
    }
    if (threadIdx.x == 0) out[0] = s[0] * invN;
}

extern "C" void kernel_launch(void* const* d_in, const int* in_sizes, int n_in,
                              void* d_out, int out_size, void* d_ws, size_t ws_size,
                              hipStream_t stream) {
    const float* x   = (const float*)d_in[0];
    const float* Wmu = (const float*)d_in[1];
    const float* bmu = (const float*)d_in[2];
    const float* Wu  = (const float*)d_in[3];
    const float* bu  = (const float*)d_in[4];
    const float* Wd  = (const float*)d_in[5];
    const float* bd  = (const float*)d_in[6];
    // d_in[7] = rank (fixed at 4; kernel is specialized)

    const int n       = in_sizes[0] / N_HIDDEN;   // 131072
    const int nblocks = n / ROWS_PB;              // 4096

    float*  partials = (float*)d_ws;                       // nblocks * 4 B
    __bf16* Wt       = (__bf16*)((char*)d_ws + 64 * 1024); // 768*512*2 B

    lrg_build_wt<<<(N_COLS * N_HIDDEN + 255) / 256, 256, 0, stream>>>(Wmu, Wu, Wd, Wt);
    lrg_fused_kl<<<nblocks, 256, 0, stream>>>(x, bmu, bu, bd, Wt, partials);
    lrg_reduce<<<1, 256, 0, stream>>>(partials, nblocks, (float*)d_out, 1.0f / (float)n);
}
